// TopologicalAttention_30133490549636
// MI455X (gfx1250) — compile-verified, matrix operand order checked by reading, not run on hardware
//
#include <hip/hip_runtime.h>

#define B_    2
#define H_    16
#define N_    2048
#define D_    64
#define BM    128
#define BN    64
#define PITCH 72          // bf16 elements per LDS row (144B) -> conflict-free frag loads
#define NTHREADS 256

typedef __bf16 bf16;
typedef __attribute__((ext_vector_type(8)))  bf16  bf16x8;
typedef __attribute__((ext_vector_type(16))) bf16  bf16x16;
typedef __attribute__((ext_vector_type(8)))  float f32x8;

// ---- VALU cross-lane butterfly within 16-lane rows via v_permlane16_b32 ----
// nibble i of {SHI,SLO} selects the source lane (within the 16-lane row) for lane i
template <unsigned SLO, unsigned SHI>
__device__ __forceinline__ float permlane16_f(float x) {
    int i = __builtin_bit_cast(int, x);
    i = __builtin_amdgcn_permlane16(i, i, (int)SLO, (int)SHI,
                                    /*fi=*/false, /*bound_ctrl=*/false);
    return __builtin_bit_cast(float, i);
}

__device__ __forceinline__ float rowmax16(float x) {
    x = fmaxf(x, permlane16_f<0x67452301u, 0xEFCDAB89u>(x)); // xor 1
    x = fmaxf(x, permlane16_f<0x54761032u, 0xDCFE98BAu>(x)); // xor 2
    x = fmaxf(x, permlane16_f<0x32107654u, 0xBA98FEDCu>(x)); // xor 4
    x = fmaxf(x, permlane16_f<0xFEDCBA98u, 0x76543210u>(x)); // xor 8
    return x;
}

__device__ __forceinline__ float rowsum16(float x) {
    x += permlane16_f<0x67452301u, 0xEFCDAB89u>(x); // xor 1
    x += permlane16_f<0x54761032u, 0xDCFE98BAu>(x); // xor 2
    x += permlane16_f<0x32107654u, 0xBA98FEDCu>(x); // xor 4
    x += permlane16_f<0xFEDCBA98u, 0x76543210u>(x); // xor 8
    return x;
}

// Load a 16x32 bf16 WMMA fragment (A or B layout, they are symmetric):
// lane = (lane&15 selects row, half = lane>>4 selects K-subset)
// VGPRs 0-3: K = 8*half .. +7 ; VGPRs 4-7: K = 16+8*half .. +7
__device__ __forceinline__ bf16x16 load_frag(const bf16* base, int rowbase,
                                             int colbase) {
    const int lane = threadIdx.x & 31;
    const bf16* p = base + (rowbase + (lane & 15)) * PITCH
                         + colbase + 8 * (lane >> 4);
    bf16x8 lo = *(const bf16x8*)(p);        // 16B aligned
    bf16x8 hi = *(const bf16x8*)(p + 16);   // 16B aligned
    return __builtin_shufflevector(lo, hi, 0,1,2,3,4,5,6,7,
                                           8,9,10,11,12,13,14,15);
}

__device__ __forceinline__ f32x8 wmma_bf16(bf16x16 a, bf16x16 b, f32x8 c) {
    return __builtin_amdgcn_wmma_f32_16x16x32_bf16(
        /*neg_a=*/false, a, /*neg_b=*/false, b,
        /*c_mod=*/(short)0, c, /*reuse_a=*/false, /*reuse_b=*/false);
}

__global__ __launch_bounds__(NTHREADS)
void topo_attn_kernel(const float* __restrict__ Q, const float* __restrict__ K,
                      const float* __restrict__ V, const float* __restrict__ top,
                      float* __restrict__ out) {
    __shared__ bf16 Qs[BM * PITCH];          // 18.4 KB, bf16 Q tile (row major)
    __shared__ bf16 Ks[BN * PITCH];          //  9.2 KB, bf16 K tile (row major)
    __shared__ bf16 Vt[D_ * PITCH];          //  9.2 KB, bf16 V tile transposed
    __shared__ bf16 Ps[8 * 16 * PITCH];      // 18.4 KB, per-wave P strips

    const int tid    = threadIdx.x;
    const int wave   = tid >> 5;
    const int lane   = tid & 31;
    const int lane16 = lane & 15;
    const int half   = lane >> 4;

    const int q0 = blockIdx.x * BM;
    const int bh = blockIdx.y;

    const size_t baseQKV = (size_t)bh * N_ * D_;
    const size_t baseTop = (size_t)bh * N_ * N_;

    // ---- stage Q tile (fp32 -> bf16) ----
    for (int i = tid; i < BM * D_; i += NTHREADS) {
        const int r = i >> 6, d = i & 63;
        Qs[r * PITCH + d] = (bf16)Q[baseQKV + (size_t)(q0 + r) * D_ + d];
    }
    __syncthreads();

    // per-wave Q A-fragments: rows wave*16..+15, K-steps d=[0,32) and [32,64)
    const bf16x16 qa0 = load_frag(Qs, wave * 16, 0);
    const bf16x16 qa1 = load_frag(Qs, wave * 16, 32);

    // online-softmax state (per lane: one value per C-layout VGPR index v;
    // row within strip = v + 8*half)
    float mrow[8], lrow[8];
    f32x8 oacc[4];
#pragma unroll
    for (int v = 0; v < 8; ++v) { mrow[v] = -1e30f; lrow[v] = 0.f; }
#pragma unroll
    for (int t = 0; t < 4; ++t) { f32x8 z = {}; oacc[t] = z; }

    const float scale = 0.125f;  // 1/sqrt(64)
    const float* topWave = top + baseTop + (size_t)(q0 + wave * 16) * N_;
    bf16* Pw = Ps + wave * 16 * PITCH;

    for (int k0 = 0; k0 < N_; k0 += BN) {
        // ---- stage K (row major) and V (transposed), fp32 -> bf16 ----
        for (int i = tid; i < BN * D_; i += NTHREADS) {
            const int r = i >> 6, d = i & 63;
            const size_t g = baseQKV + (size_t)(k0 + r) * D_ + d;
            Ks[r * PITCH + d] = (bf16)K[g];
            Vt[d * PITCH + r] = (bf16)V[g];
        }
        // prefetch next bias tile rows into L2 (bandwidth-critical stream)
        if (k0 + BN < N_)
            __builtin_prefetch(topWave + (size_t)lane16 * N_ + k0 + BN + half * 32, 0, 1);
        __syncthreads();

        // ---- S = Q * K^T (4 n-tiles x 2 K-steps = 8 WMMAs) ----
        f32x8 s[4];
#pragma unroll
        for (int t = 0; t < 4; ++t) {
            f32x8 z = {};
            z = wmma_bf16(qa0, load_frag(Ks, t * 16, 0), z);
            z = wmma_bf16(qa1, load_frag(Ks, t * 16, 32), z);
            s[t] = z;
        }

        // ---- scale + bias (streamed from HBM, C-layout addressed) ----
#pragma unroll
        for (int t = 0; t < 4; ++t)
#pragma unroll
            for (int v = 0; v < 8; ++v) {
                const int r = v + 8 * half;
                s[t][v] = s[t][v] * scale
                        + topWave[(size_t)r * N_ + k0 + t * 16 + lane16];
            }

        // ---- online softmax: row max / exp / row sum (VALU permlane16) ----
        float mnew[8], alpha[8], rsum[8];
#pragma unroll
        for (int v = 0; v < 8; ++v) {
            float mv = fmaxf(fmaxf(s[0][v], s[1][v]), fmaxf(s[2][v], s[3][v]));
            mv       = rowmax16(mv);
            mnew[v]  = fmaxf(mrow[v], mv);
            alpha[v] = __expf(mrow[v] - mnew[v]);
            mrow[v]  = mnew[v];
            rsum[v]  = 0.f;
        }
#pragma unroll
        for (int t = 0; t < 4; ++t)
#pragma unroll
            for (int v = 0; v < 8; ++v) {
                const float p = __expf(s[t][v] - mnew[v]);
                s[t][v] = p;
                rsum[v] += p;
            }
#pragma unroll
        for (int v = 0; v < 8; ++v)
            lrow[v] = lrow[v] * alpha[v] + rowsum16(rsum[v]);
#pragma unroll
        for (int t = 0; t < 4; ++t)
#pragma unroll
            for (int v = 0; v < 8; ++v)
                oacc[t][v] *= alpha[v];

        // ---- P (bf16) -> per-wave LDS strip (C layout -> row major) ----
#pragma unroll
        for (int t = 0; t < 4; ++t)
#pragma unroll
            for (int v = 0; v < 8; ++v)
                Pw[(v + 8 * half) * PITCH + t * 16 + lane16] = (bf16)s[t][v];

        // ---- O += P * V (4 d-tiles x 2 K-steps = 8 WMMAs) ----
        const bf16x16 pa0 = load_frag(Pw, 0, 0);
        const bf16x16 pa1 = load_frag(Pw, 0, 32);
#pragma unroll
        for (int t = 0; t < 4; ++t) {
            oacc[t] = wmma_bf16(pa0, load_frag(Vt, t * 16, 0), oacc[t]);
            oacc[t] = wmma_bf16(pa1, load_frag(Vt, t * 16, 32), oacc[t]);
        }
        __syncthreads();
    }

    // ---- epilogue: normalize (one rcp per row) and store fp32 ----
    float rinv[8];
#pragma unroll
    for (int v = 0; v < 8; ++v) rinv[v] = 1.0f / lrow[v];

    float* outWave = out + baseQKV + (size_t)(q0 + wave * 16) * D_;
#pragma unroll
    for (int t = 0; t < 4; ++t)
#pragma unroll
        for (int v = 0; v < 8; ++v) {
            const int r = v + 8 * half;
            outWave[(size_t)r * D_ + t * 16 + lane16] = oacc[t][v] * rinv[v];
        }
}

extern "C" void kernel_launch(void* const* d_in, const int* in_sizes, int n_in,
                              void* d_out, int out_size, void* d_ws, size_t ws_size,
                              hipStream_t stream) {
    const float* Q   = (const float*)d_in[0];
    const float* K   = (const float*)d_in[1];
    const float* V   = (const float*)d_in[2];
    const float* top = (const float*)d_in[3];
    float* out       = (float*)d_out;

    dim3 grid(N_ / BM, B_ * H_);
    topo_attn_kernel<<<grid, NTHREADS, 0, stream>>>(Q, K, V, top, out);
}
